// DeformableTransformerDecoderLayer_20693152432679
// MI455X (gfx1250) — matrix operand order repaired and checked by reading, not compiled, not run on hardware
//
#include <hip/hip_runtime.h>
#include <hip/hip_bf16.h>
#include <math.h>

// ---------------- problem constants (from reference) ----------------
#define T_   6
#define LQ_  300
#define C_   256
#define NH_  8
#define NL_  4
#define PC_  4
#define TW_  2
#define PT_  2
#define DFF_ 1024
#define HD_  32
#define LIN_ 16320            // 96*128 + 48*64 + 24*32 + 12*16
#define ROWS_ (T_ * LQ_)      // 1800

// ---------------- vector types ----------------
typedef __attribute__((ext_vector_type(16))) __bf16         bf16x16;
typedef __attribute__((ext_vector_type(8)))  float          f32x8;
typedef __attribute__((ext_vector_type(16))) unsigned short u16x16;
typedef __attribute__((ext_vector_type(4)))  unsigned int   u32x4;
typedef __attribute__((ext_vector_type(8)))  int            i32x8;
typedef __attribute__((ext_vector_type(4)))  int            i32x4;

#if __has_builtin(__builtin_amdgcn_tensor_load_to_lds) && __has_builtin(__builtin_amdgcn_s_wait_tensorcnt)
#define HAVE_TDM 1
#else
#define HAVE_TDM 0
#endif

// round-to-nearest-even f32 -> bf16 bit pattern
__device__ __forceinline__ unsigned short f32_bf16(float f) {
    unsigned int u = __float_as_uint(f);
    unsigned int r = u + 0x7FFFu + ((u >> 16) & 1u);
    return (unsigned short)(r >> 16);
}

// A-fragment layout within each 32-element K-chunk so a lane's 16 values are
// contiguous 32B.  Lane-half h holds K = {16h..16h+15} (same order as the B-frag).
// Position of K-index k5 (0..31) inside the stored chunk:
__device__ __forceinline__ int swzpos(int k5) {
    return k5;
}
__device__ __forceinline__ size_t swzcol(size_t base, int col) {
    return base + (size_t)(col & ~31) + swzpos(col & 31);
}

__device__ __forceinline__ float wred_sum(float v) {
#pragma unroll
    for (int m = 16; m > 0; m >>= 1) v += __shfl_xor(v, m, 32);
    return v;
}
__device__ __forceinline__ float wred_max(float v) {
#pragma unroll
    for (int m = 16; m > 0; m >>= 1) v = fmaxf(v, __shfl_xor(v, m, 32));
    return v;
}

// ---------------- elementwise conversion kernels ----------------
// weights: plain [N,K] bf16 (B-fragments want contiguous K -> no swizzle)
__global__ void cvt_bf16_kernel(const float* __restrict__ s,
                                unsigned short* __restrict__ d, int n) {
    int i = blockIdx.x * blockDim.x + threadIdx.x;
    if (i < n) d[i] = f32_bf16(s[i]);
}
// activations: bf16 + A-fragment swizzle (K % 32 == 0 and rows 32-aligned here)
__global__ void cvt_swz_kernel(const float* __restrict__ s,
                               unsigned short* __restrict__ d, int n) {
    int i = blockIdx.x * blockDim.x + threadIdx.x;
    if (i < n) d[(i & ~31) + swzpos(i & 31)] = f32_bf16(s[i]);
}
__global__ void add_cvt_swz_kernel(const float* __restrict__ a, const float* __restrict__ b,
                                   unsigned short* __restrict__ d, int n) {
    int i = blockIdx.x * blockDim.x + threadIdx.x;
    if (i < n) d[(i & ~31) + swzpos(i & 31)] = f32_bf16(a[i] + b[i]);
}

// ---------------- TDM: stage a [64 x 256] bf16 weight panel into LDS ----------------
// LDS layout: 64 rows x 272 ushorts (256 data + 16 pad -> 544B stride, 32B aligned,
// rows shifted by 8 banks).  Padding produced by the TDM descriptor itself:
// pad_interval code 6 = 128 DWORDs (=256 elems = one row), pad_amount code 7 = 8 DWORDs (=16 ushorts).
#define BROW_ 272
#define BPANEL_ (64 * BROW_)

#if HAVE_TDM
__device__ __forceinline__ void tdm_load_b_panel(const unsigned short* gsrc,
                                                 unsigned int lds_off, int rowStrideElems) {
    unsigned long long ga = (unsigned long long)(uintptr_t)gsrc;
    u32x4 g0;
    g0[0] = 1u;                                              // count=1, user descriptor
    g0[1] = lds_off;                                         // lds_addr (bytes)
    g0[2] = (unsigned int)(ga & 0xFFFFFFFFu);                // global_addr[31:0]
    g0[3] = (unsigned int)((ga >> 32) & 0x01FFFFFFu) | (2u << 30);  // addr[56:32] | type=2
    const unsigned int dw0 = (1u << 16)   // data_size = 2 bytes
                           | (1u << 20)   // pad_enable
                           | (6u << 22)   // pad_interval: 128 DW before pad (= 256 elems)
                           | (7u << 25);  // pad_amount: 8 DW (= 16 ushorts)
    const unsigned int td0 = 256u, td1 = 64u;                // tensor dims (elems)
    const unsigned int tile0 = 256u, tile1 = 64u;            // tile dims
    const unsigned long long s0 = (unsigned long long)rowStrideElems;  // dim0 stride
    i32x8 g1;
    g1[0] = (int)dw0;
    g1[1] = (int)((td0 & 0xFFFFu) << 16);                    // tensor_dim0[15:0] @ bit48
    g1[2] = (int)((td0 >> 16) | ((td1 & 0xFFFFu) << 16));    // td0[31:16] | td1[15:0]
    g1[3] = (int)((td1 >> 16) | (tile0 << 16));              // td1[31:16] | tile_dim0
    g1[4] = (int)(tile1 & 0xFFFFu);                          // tile_dim1 | tile_dim2=0
    g1[5] = (int)(unsigned int)(s0 & 0xFFFFFFFFu);           // dim0_stride[31:0]
    g1[6] = (int)(unsigned int)((s0 >> 32) & 0xFFFFu);       // dim0_stride[47:32] | dim1_stride[15:0]=0
    g1[7] = 0;
    i32x4 g2 = {0, 0, 0, 0};
    i32x4 g3 = {0, 0, 0, 0};
#if __clang_major__ >= 23
    i32x8 g4 = {0, 0, 0, 0, 0, 0, 0, 0};
    __builtin_amdgcn_tensor_load_to_lds(g0, g1, g2, g3, g4, 0);
#else
    __builtin_amdgcn_tensor_load_to_lds(g0, g1, g2, g3, 0);
#endif
}
#endif

// ---------------- WMMA GEMM ----------------
//   C[M,N] = act( (A[M,K] * W[N,K]^T + bias) * alpha )   (outputs fp32 and/or swizzled bf16)
// A: bf16, pre-swizzled so a lane's A-fragment is one contiguous 32B load.
// W: bf16 [N,K].  Block = 256 rows x 64 cols; 8 waves, each computing 32x64
// (2 A-frags x 4 B-frags -> 8 v_wmma per 32-K step, register B reuse).
// B panel staged in LDS by the Tensor Data Mover in 64x256 chunks.
template <int KK>
__global__ __launch_bounds__(256)
void gemm_bf16_wmma(const unsigned short* __restrict__ Abf,
                    const unsigned short* __restrict__ Wb,
                    const float* __restrict__ bias,
                    float* __restrict__ Cf, unsigned short* __restrict__ Cb,
                    int M, int N, float alpha, int relu,
                    const unsigned char* __restrict__ row_mask) {
    __shared__ __align__(64) unsigned short Bs[BPANEL_];
    const int nt64 = N >> 6;
    const int bm = blockIdx.x / nt64, bn = blockIdx.x % nt64;
    const int n0 = bn << 6;
    const int wid  = threadIdx.x >> 5;
    const int lane = threadIdx.x & 31;
    const int half = lane >> 4, l16 = lane & 15;
    const int m0 = bm * 256 + wid * 32;
    int mr0 = m0 + l16;      if (mr0 >= M) mr0 = M - 1;   // clamp loads, mask stores
    int mr1 = m0 + 16 + l16; if (mr1 >= M) mr1 = M - 1;
    const unsigned short* A0 = Abf + (size_t)mr0 * KK;
    const unsigned short* A1 = Abf + (size_t)mr1 * KK;

    f32x8 acc[2][4];
#pragma unroll
    for (int i = 0; i < 2; ++i)
#pragma unroll
        for (int j = 0; j < 4; ++j) acc[i][j] = f32x8{};

    for (int ko = 0; ko < KK; ko += 256) {
#if HAVE_TDM
        if (wid == 0) {            // one wave issues the TDM transfer for the block
            tdm_load_b_panel(Wb + (size_t)n0 * KK + ko,
                             (unsigned int)(uintptr_t)(void*)Bs, KK);
            __builtin_amdgcn_s_wait_tensorcnt(0);
        }
#else
        for (int idx = threadIdx.x; idx < 64 * 256; idx += 256) {
            const int rr = idx >> 8, kk = idx & 255;
            Bs[rr * BROW_ + kk] = Wb[(size_t)(n0 + rr) * KK + ko + kk];
        }
#endif
        __syncthreads();
#pragma unroll
        for (int k0 = 0; k0 < 256; k0 += 32) {
            const u16x16 au0 = *reinterpret_cast<const u16x16*>(A0 + ko + k0 + 16 * half);
            const u16x16 au1 = *reinterpret_cast<const u16x16*>(A1 + ko + k0 + 16 * half);
            const bf16x16 af0 = __builtin_bit_cast(bf16x16, au0);
            const bf16x16 af1 = __builtin_bit_cast(bf16x16, au1);
#pragma unroll
            for (int j = 0; j < 4; ++j) {
                // B-frag: lane-half h holds K = k0+16h..k0+16h+15 of column n0+j*16+l16
                const u16x16 bu = *reinterpret_cast<const u16x16*>(
                    &Bs[(j * 16 + l16) * BROW_ + k0 + 16 * half]);
                const bf16x16 bf = __builtin_bit_cast(bf16x16, bu);
                acc[0][j] = __builtin_amdgcn_wmma_f32_16x16x32_bf16(
                    false, af0, false, bf, (short)0, acc[0][j], false, false);
                acc[1][j] = __builtin_amdgcn_wmma_f32_16x16x32_bf16(
                    false, af1, false, bf, (short)0, acc[1][j], false, false);
            }
        }
        __syncthreads();
    }

    // ---- epilogue: bias, scale, relu, row mask; fp32 and/or swizzled-bf16 store ----
#pragma unroll
    for (int i = 0; i < 2; ++i)
#pragma unroll
        for (int j = 0; j < 4; ++j) {
            const int col = n0 + j * 16 + l16;
            const float bv = bias ? bias[col] : 0.f;
#pragma unroll
            for (int r = 0; r < 8; ++r) {
                const int row = m0 + i * 16 + half * 8 + r;  // C/D layout: VGPR r, half -> M+8
                if (row < M) {
                    float v = (acc[i][j][r] + bv) * alpha;
                    if (relu) v = fmaxf(v, 0.f);
                    if (row_mask && row_mask[row]) v = 0.f;
                    if (Cf) Cf[(size_t)row * N + col] = v;
                    if (Cb) Cb[swzcol((size_t)row * N, col)] = f32_bf16(v);
                }
            }
        }
}

// ---------------- self-attention: one wave per (t,h,q); lane = d (HD==32==wave32) ----------------
__global__ __launch_bounds__(256)
void self_attn_kernel(const float* __restrict__ qh, const float* __restrict__ kh,
                      const float* __restrict__ vh, unsigned short* __restrict__ outb) {
    __shared__ float sc[NH_][LQ_ + 4];
    const int t = blockIdx.x / LQ_;
    const int q = blockIdx.x % LQ_;
    const int h = threadIdx.x >> 5;
    const int lane = threadIdx.x & 31;

    const float qd = qh[(size_t)(t * LQ_ + q) * C_ + h * HD_ + lane];  // pre-scaled 1/sqrt(HD)
    const float* kbase = kh + (size_t)t * LQ_ * C_ + h * HD_;
    for (int j = 0; j < LQ_; ++j) {
        float s = wred_sum(qd * kbase[(size_t)j * C_ + lane]);
        if (lane == 0) sc[h][j] = s;
    }
    __syncthreads();

    float mx = -1e30f;
    for (int j = lane; j < LQ_; j += 32) mx = fmaxf(mx, sc[h][j]);
    mx = wred_max(mx);
    float sum = 0.f;
    for (int j = lane; j < LQ_; j += 32) { float e = __expf(sc[h][j] - mx); sc[h][j] = e; sum += e; }
    sum = wred_sum(sum);
    const float inv = 1.f / sum;
    __syncthreads();

    const float* vbase = vh + (size_t)t * LQ_ * C_ + h * HD_;
    float o = 0.f;
    for (int j = 0; j < LQ_; ++j) o = fmaf(sc[h][j], vbase[(size_t)j * C_ + lane], o);
    // bf16 + A-swizzle (chunk == head since HD == 32)
    outb[(size_t)(t * LQ_ + q) * C_ + h * HD_ + swzpos(lane)] = f32_bf16(o * inv);
}

// ---------------- layernorm over C=256 ----------------
// out = LN(res+delta)*g+b (fp32, may be d_out); optional swizzled-bf16 copy; optional
// swizzled-bf16 of (out + extra).
__global__ __launch_bounds__(256)
void layernorm_kernel(const float* __restrict__ res, const float* __restrict__ delta,
                      const float* __restrict__ g, const float* __restrict__ b,
                      float* __restrict__ out, unsigned short* __restrict__ out_bf,
                      const float* __restrict__ extra, unsigned short* __restrict__ out2_bf) {
    __shared__ float red[8];
    __shared__ float stat[2];
    const int row = blockIdx.x, tid = threadIdx.x;
    const int wid = tid >> 5, lane = tid & 31;
    const size_t base = (size_t)row * C_;

    const float x = res[base + tid] + delta[base + tid];
    float s = wred_sum(x);
    if (lane == 0) red[wid] = s;
    __syncthreads();
    if (tid == 0) { float tot = 0; for (int i = 0; i < 8; ++i) tot += red[i]; stat[0] = tot / C_; }
    __syncthreads();
    const float mean = stat[0];
    const float dx = x - mean;
    float s2 = wred_sum(dx * dx);
    __syncthreads();
    if (lane == 0) red[wid] = s2;
    __syncthreads();
    if (tid == 0) { float tot = 0; for (int i = 0; i < 8; ++i) tot += red[i]; stat[1] = rsqrtf(tot / C_ + 1e-5f); }
    __syncthreads();
    const float y = dx * stat[1] * g[tid] + b[tid];
    out[base + tid] = y;
    if (out_bf)  out_bf[swzcol(base, tid)]  = f32_bf16(y);
    if (out2_bf) out2_bf[swzcol(base, tid)] = f32_bf16(y + extra[base + tid]);
}

// ---------------- softmax over 32 attn weights per (t,q,h): wave per row ----------------
__global__ __launch_bounds__(256)
void aw_softmax_kernel(float* __restrict__ aw) {
    const int row = blockIdx.x;                 // 0..1799
    const int h = threadIdx.x >> 5, lane = threadIdx.x & 31;
    const size_t idx = (size_t)row * C_ + h * 32 + lane;
    const float v = aw[idx];
    const float mx = wred_max(v);
    const float e = __expf(v - mx);
    const float s = wred_sum(e);
    aw[idx] = e / s;
}

// ---------------- deformable sampling ----------------
__constant__ const int c_HL[NL_] = {96, 48, 24, 12};
__constant__ const int c_WL[NL_] = {128, 64, 32, 16};
__constant__ const int c_ST[NL_] = {0, 12288, 15360, 16128};

__device__ __forceinline__ float bilin(const float* __restrict__ value, int frame, int st,
                                       int hl, int wl, int h, int d, float lx, float ly) {
    const float x = lx * wl - 0.5f, y = ly * hl - 0.5f;
    const float x0f = floorf(x), y0f = floorf(y);
    const int x0 = (int)x0f, y0 = (int)y0f;
    const float wx = x - x0f, wy = y - y0f;
    float r = 0.f;
#pragma unroll
    for (int cy = 0; cy < 2; ++cy)
#pragma unroll
        for (int cx = 0; cx < 2; ++cx) {
            const int xi = x0 + cx, yi = y0 + cy;
            if (xi >= 0 && xi < wl && yi >= 0 && yi < hl) {
                const float w = (cx ? wx : 1.f - wx) * (cy ? wy : 1.f - wy);
                const size_t vidx = ((size_t)(frame * LIN_ + st + yi * wl + xi) * NH_ + h) * HD_ + d;
                r = fmaf(w, value[vidx], r);   // coalesced: lane == d
            }
        }
    return r;
}

__global__ __launch_bounds__(256)
void deform_kernel(const float* __restrict__ value, const float* __restrict__ refpts,
                   const float* __restrict__ cfo, const float* __restrict__ ofo,
                   const float* __restrict__ soff, const float* __restrict__ toff,
                   const float* __restrict__ aw, unsigned short* __restrict__ outb) {
    const int t = blockIdx.x / LQ_;
    const int q = blockIdx.x % LQ_;
    const int h = threadIdx.x >> 5;       // wave == head
    const int d = threadIdx.x & 31;       // lane == channel
    const int row = t * LQ_ + q;

    float acc = 0.f;
#pragma unroll
    for (int l = 0; l < NL_; ++l) {
        const int hl = c_HL[l], wl = c_WL[l], st = c_ST[l];
        const float inv_w = 1.f / (float)wl, inv_h = 1.f / (float)hl;
        const float rx = refpts[(size_t)(row * NL_ + l) * 2 + 0];
        const float ry = refpts[(size_t)(row * NL_ + l) * 2 + 1];

        const float cx = cfo[(size_t)(row * NL_ + l) * 2 + 0];
        const float cy = cfo[(size_t)(row * NL_ + l) * 2 + 1];
#pragma unroll
        for (int p = 0; p < PC_; ++p) {
            const size_t so = ((size_t)(row * NH_ + h) * NL_ + l) * PC_ + p;
            const float lx = rx + cx + soff[so * 2 + 0] * inv_w;
            const float ly = ry + cy + soff[so * 2 + 1] * inv_h;
            const float a = aw[(size_t)(row * NH_ + h) * 32 + l * 8 + p];
            acc = fmaf(a, bilin(value, t, st, hl, wl, h, d, lx, ly), acc);
        }
#pragma unroll
        for (int w = 0; w < TW_; ++w) {           // temporal deltas {-1,+1}
            int frame = t + (w ? 1 : -1);
            frame = frame < 0 ? 0 : (frame > T_ - 1 ? T_ - 1 : frame);
            const float ox0 = ofo[(((size_t)row * TW_ + w) * NL_ + l) * 2 + 0];
            const float oy0 = ofo[(((size_t)row * TW_ + w) * NL_ + l) * 2 + 1];
#pragma unroll
            for (int p = 0; p < PT_; ++p) {
                const size_t to = (((size_t)(row * NH_ + h) * NL_ + l) * TW_ + w) * PT_ + p;
                const float lx = rx + ox0 + toff[to * 2 + 0] * inv_w;
                const float ly = ry + oy0 + toff[to * 2 + 1] * inv_h;
                const float a = aw[(size_t)(row * NH_ + h) * 32 + l * 8 + PC_ + w * PT_ + p];
                acc = fmaf(a, bilin(value, frame, st, hl, wl, h, d, lx, ly), acc);
            }
        }
    }
    outb[(size_t)row * C_ + h * HD_ + swzpos(d)] = f32_bf16(acc);   // bf16 + A-swizzle
}

// ---------------- host orchestration ----------------
extern "C" void kernel_launch(void* const* d_in, const int* in_sizes, int n_in,
                              void* d_out, int out_size, void* d_ws, size_t ws_size,
                              hipStream_t stream) {
    (void)in_sizes; (void)n_in; (void)out_size; (void)ws_size;

    const float* tgt   = (const float*)d_in[0];
    const float* qpos  = (const float*)d_in[1];
    const float* refp  = (const float*)d_in[2];
    const float* cfo   = (const float*)d_in[3];
    const float* ofo   = (const float*)d_in[4];
    const float* src   = (const float*)d_in[5];
    // d_in[6] src_spatial_shapes, d_in[7] level_start_index: compile-time constants here
    const unsigned char* mask = (const unsigned char*)d_in[8];
    const float* in_proj_w = (const float*)d_in[9];
    const float* in_proj_b = (const float*)d_in[10];
    const float* sa_out_w  = (const float*)d_in[11];
    const float* sa_out_b  = (const float*)d_in[12];
    const float* n1g = (const float*)d_in[13], *n1b = (const float*)d_in[14];
    const float* n2g = (const float*)d_in[15], *n2b = (const float*)d_in[16];
    const float* n3g = (const float*)d_in[17], *n3b = (const float*)d_in[18];
    const float* value_w = (const float*)d_in[19], *value_b = (const float*)d_in[20];
    const float* soff_w  = (const float*)d_in[21], *soff_b  = (const float*)d_in[22];
    const float* toff_w  = (const float*)d_in[23], *toff_b  = (const float*)d_in[24];
    const float* attn_w  = (const float*)d_in[25], *attn_b  = (const float*)d_in[26];
    const float* cross_w = (const float*)d_in[27], *cross_b = (const float*)d_in[28];
    const float* lin1_w  = (const float*)d_in[29], *lin1_b  = (const float*)d_in[30];
    const float* lin2_w  = (const float*)d_in[31], *lin2_b  = (const float*)d_in[32];

    // bump allocator over d_ws (~185 MB total)
    char* p = (char*)d_ws;
    auto alloc = [&](size_t bytes) -> void* {
        void* r = (void*)p;
        p += (bytes + 255) & ~(size_t)255;
        return r;
    };
    // bf16 weights, [N,K] layout
    unsigned short* wb_inproj = (unsigned short*)alloc(3 * C_ * C_ * 2);
    unsigned short* wb_saout  = (unsigned short*)alloc(C_ * C_ * 2);
    unsigned short* wb_value  = (unsigned short*)alloc(C_ * C_ * 2);
    unsigned short* wb_soff   = (unsigned short*)alloc(C_ * C_ * 2);
    unsigned short* wb_toff   = (unsigned short*)alloc(C_ * C_ * 2);
    unsigned short* wb_attn   = (unsigned short*)alloc(C_ * C_ * 2);
    unsigned short* wb_cross  = (unsigned short*)alloc(C_ * C_ * 2);
    unsigned short* wb_lin1   = (unsigned short*)alloc(DFF_ * C_ * 2);
    unsigned short* wb_lin2   = (unsigned short*)alloc(C_ * DFF_ * 2);

    const size_t RC = (size_t)ROWS_ * C_;                 // 460800
    // bf16 activations (A-swizzled)
    unsigned short* qbuf_bf = (unsigned short*)alloc(RC * 2);
    unsigned short* tgtA_bf = (unsigned short*)alloc(RC * 2);
    unsigned short* sain_bf = (unsigned short*)alloc(RC * 2);
    unsigned short* qc_bf   = (unsigned short*)alloc(RC * 2);
    unsigned short* dout_bf = (unsigned short*)alloc(RC * 2);
    unsigned short* tgt3_bf = (unsigned short*)alloc(RC * 2);
    unsigned short* ffh_bf  = (unsigned short*)alloc((size_t)ROWS_ * DFF_ * 2);
    unsigned short* src_bf  = (unsigned short*)alloc((size_t)T_ * LIN_ * C_ * 2);   // 50 MB
    // fp32 buffers
    float* qh    = (float*)alloc(RC * 4);
    float* kh    = (float*)alloc(RC * 4);
    float* vh    = (float*)alloc(RC * 4);
    float* sa    = (float*)alloc(RC * 4);
    float* tgt2  = (float*)alloc(RC * 4);
    float* soff  = (float*)alloc(RC * 4);
    float* toff  = (float*)alloc(RC * 4);
    float* aw    = (float*)alloc(RC * 4);
    float* cross = (float*)alloc(RC * 4);
    float* tgt3  = (float*)alloc(RC * 4);
    float* ff    = (float*)alloc(RC * 4);
    float* value = (float*)alloc((size_t)T_ * LIN_ * C_ * 4);   // 100.3 MB

    auto cvt = [&](const float* s, unsigned short* d, int n) {
        cvt_bf16_kernel<<<(n + 255) / 256, 256, 0, stream>>>(s, d, n);
    };
    auto cvt_swz = [&](const float* s, unsigned short* d, int n) {
        cvt_swz_kernel<<<(n + 255) / 256, 256, 0, stream>>>(s, d, n);
    };
    auto gemm = [&](const unsigned short* A, const unsigned short* Wb, const float* bias,
                    float* Cf, unsigned short* Cb, int M, int N, int K,
                    float alpha, int relu, const unsigned char* rm) {
        const int blocks = ((M + 255) / 256) * (N / 64);
        if (K == 256)
            gemm_bf16_wmma<256><<<blocks, 256, 0, stream>>>(A, Wb, bias, Cf, Cb, M, N, alpha, relu, rm);
        else
            gemm_bf16_wmma<1024><<<blocks, 256, 0, stream>>>(A, Wb, bias, Cf, Cb, M, N, alpha, relu, rm);
    };

    // ---- one-time conversions ----
    cvt(in_proj_w, wb_inproj, 3 * C_ * C_);
    cvt(sa_out_w,  wb_saout,  C_ * C_);
    cvt(value_w,   wb_value,  C_ * C_);
    cvt(soff_w,    wb_soff,   C_ * C_);
    cvt(toff_w,    wb_toff,   C_ * C_);
    cvt(attn_w,    wb_attn,   C_ * C_);
    cvt(cross_w,   wb_cross,  C_ * C_);
    cvt(lin1_w,    wb_lin1,   DFF_ * C_);
    cvt(lin2_w,    wb_lin2,   C_ * DFF_);
    cvt_swz(tgt, tgtA_bf, (int)RC);
    cvt_swz(src, src_bf, T_ * LIN_ * C_);
    add_cvt_swz_kernel<<<(int)((RC + 255) / 256), 256, 0, stream>>>(tgt, qpos, qbuf_bf, (int)RC);

    // ---- self-attention block ----
    const float inv_sqrt_hd = 0.17677669529663687f;      // 1/sqrt(32)
    gemm(qbuf_bf, wb_inproj + 0 * C_ * C_, in_proj_b + 0 * C_, qh, nullptr, ROWS_, C_, C_, inv_sqrt_hd, 0, nullptr);
    gemm(qbuf_bf, wb_inproj + 1 * C_ * C_, in_proj_b + 1 * C_, kh, nullptr, ROWS_, C_, C_, 1.f, 0, nullptr);
    gemm(tgtA_bf, wb_inproj + 2 * C_ * C_, in_proj_b + 2 * C_, vh, nullptr, ROWS_, C_, C_, 1.f, 0, nullptr);
    self_attn_kernel<<<ROWS_, 256, 0, stream>>>(qh, kh, vh, sain_bf);
    gemm(sain_bf, wb_saout, sa_out_b, sa, nullptr, ROWS_, C_, C_, 1.f, 0, nullptr);
    // tgt2 = LN(tgt + sa; norm2) ; qc_bf = bf16(tgt2 + query_pos)
    layernorm_kernel<<<ROWS_, 256, 0, stream>>>(tgt, sa, n2g, n2b, tgt2, nullptr, qpos, qc_bf);

    // ---- cross (deformable) attention block ----
    gemm(src_bf, wb_value, value_b, value, nullptr, T_ * LIN_, C_, C_, 1.f, 0, mask);  // dominant GEMM
    gemm(qc_bf, wb_soff, soff_b, soff, nullptr, ROWS_, C_, C_, 1.f, 0, nullptr);
    gemm(qc_bf, wb_toff, toff_b, toff, nullptr, ROWS_, C_, C_, 1.f, 0, nullptr);
    gemm(qc_bf, wb_attn, attn_b, aw,   nullptr, ROWS_, C_, C_, 1.f, 0, nullptr);
    aw_softmax_kernel<<<ROWS_, 256, 0, stream>>>(aw);
    deform_kernel<<<ROWS_, 256, 0, stream>>>(value, refp, cfo, ofo, soff, toff, aw, dout_bf);
    gemm(dout_bf, wb_cross, cross_b, cross, nullptr, ROWS_, C_, C_, 1.f, 0, nullptr);
    // tgt3 = LN(tgt2 + cross; norm1), plus bf16 copy for FFN
    layernorm_kernel<<<ROWS_, 256, 0, stream>>>(tgt2, cross, n1g, n1b, tgt3, tgt3_bf, nullptr, nullptr);

    // ---- FFN ----
    gemm(tgt3_bf, wb_lin1, lin1_b, nullptr, ffh_bf, ROWS_, DFF_, C_, 1.f, 1, nullptr);  // relu, bf16 out
    gemm(ffh_bf,  wb_lin2, lin2_b, ff, nullptr, ROWS_, C_, DFF_, 1.f, 0, nullptr);
    // out = LN(tgt3 + ff; norm3) -> d_out (fp32)
    layernorm_kernel<<<ROWS_, 256, 0, stream>>>(tgt3, ff, n3g, n3b, (float*)d_out, nullptr, nullptr, nullptr);
}
